// ContextLayer_22935125361195
// MI455X (gfx1250) — matrix operand order repaired and checked by reading, not compiled, not run on hardware
//
#include <hip/hip_runtime.h>

// ---------------------------------------------------------------------------
// ContextLayer for MI455X (gfx1250, wave32).
//
//   g0 = X @ W0^T + b0 ; g1 = X @ W1^T + b1          (WMMA bf16x3, fp32 acc)
//   out[b,q,d] = tanh( m_q/len_b * sum_k x[b,k,d]*sigmoid(g0[b,q,d]+g1[b,k,d])*m_k )
//
// All working data (~6.5MB) is L2-resident. GEMMs run on the matrix pipe via
// v_wmma_f32_16x16x32_bf16 with a hi/lo bf16 split of fp32 operands (3 WMMAs
// per k-step) for fp32-class accuracy. The nonlinear pairwise reduction
// (105M sigmoids) is trans-pipe bound; q-tiling (QT=8) amortizes each x/g1
// load across 8 sigmoid+FMA ops so L2 traffic (~128MB) stays under the
// trans-pipe time instead of 8x above it.
// ---------------------------------------------------------------------------

typedef __bf16 bf16_t;
typedef __attribute__((ext_vector_type(8)))  __bf16 v8bf;
typedef __attribute__((ext_vector_type(16))) __bf16 v16bf;
typedef __attribute__((ext_vector_type(8)))  float  v8f;

#define B_DIM   4
#define S_DIM   256
#define D_DIM   400          // feature dim
#define KPAD    416          // D padded to multiple of 32 (WMMA K)
#define BQ      (B_DIM * S_DIM)   // 1024 rows
#define MT      (BQ / 16)    // 64 M-tiles
#define NT      (D_DIM / 16) // 25 N-tiles
#define KT      (KPAD / 32)  // 13 K-steps
#define QT      8            // query positions per block in context_reduce

// ---- split fp32 -> bf16 hi/lo, pad K to 416, row-major [BQ][KPAD] ----------
__global__ __launch_bounds__(256)
void cvt_split_x(const float* __restrict__ X,
                 bf16_t* __restrict__ Xh, bf16_t* __restrict__ Xl) {
    int idx = blockIdx.x * 256 + threadIdx.x;
    if (idx >= BQ * KPAD) return;
    int r = idx / KPAD, k = idx - r * KPAD;
    float v = (k < D_DIM) ? X[r * D_DIM + k] : 0.0f;
    bf16_t h = (bf16_t)v;
    bf16_t l = (bf16_t)(v - (float)h);
    Xh[idx] = h;  Xl[idx] = l;
}

// ---- W[e][d] -> W^T hi/lo, layout [k=KPAD][n=D_DIM] (k-major for B frags) --
__global__ __launch_bounds__(256)
void cvt_split_wT(const float* __restrict__ W,
                  bf16_t* __restrict__ Bh, bf16_t* __restrict__ Bl) {
    int idx = blockIdx.x * 256 + threadIdx.x;
    if (idx >= KPAD * D_DIM) return;
    int k = idx / D_DIM, n = idx - k * D_DIM;
    float v = (k < D_DIM) ? W[n * D_DIM + k] : 0.0f;
    bf16_t h = (bf16_t)v;
    bf16_t l = (bf16_t)(v - (float)h);
    Bh[idx] = h;  Bl[idx] = l;
}

// ---- sentence lengths: len[b] = sum_s masks[b,s] ---------------------------
__global__ __launch_bounds__(256)
void seq_lengths(const int* __restrict__ masks, float* __restrict__ lens) {
    __shared__ float sm[256];
    int b = blockIdx.x;
    sm[threadIdx.x] = (float)masks[b * S_DIM + threadIdx.x];
    __syncthreads();
    for (int s = 128; s > 0; s >>= 1) {
        if (threadIdx.x < s) sm[threadIdx.x] += sm[threadIdx.x + s];
        __syncthreads();
    }
    if (threadIdx.x == 0) lens[b] = sm[0];
}

// ---- C = A @ B + bias, bf16x3 via v_wmma_f32_16x16x32_bf16 -----------------
// A: [BQ][KPAD] row-major hi/lo. B: [KPAD][D_DIM] (= W^T) hi/lo. C: [BQ][D_DIM] f32.
// One wave per 16x16 C tile; 8 waves / block; 200 blocks cover 1600 tiles.
__global__ __launch_bounds__(256)
void gemm_bf16x3(const bf16_t* __restrict__ Ah, const bf16_t* __restrict__ Al,
                 const bf16_t* __restrict__ Bh, const bf16_t* __restrict__ Bl,
                 const float* __restrict__ bias, float* __restrict__ C) {
    int wave = threadIdx.x >> 5;
    int lane = threadIdx.x & 31;
    int tile = blockIdx.x * 8 + wave;
    if (tile >= MT * NT) return;          // whole-wave exit: EXEC stays all-1s
    int mt = tile / NT, nt = tile - mt * NT;

    int m16   = lane & 15;                // row within tile for A, col for C
    int khalf = lane >> 4;                // which K-half this lane holds (A frag)
    int rowA  = mt * 16 + m16;

    float bv = bias[nt * 16 + m16];       // C layout: N = lane%16 -> bias per lane
    v8f acc = {bv, bv, bv, bv, bv, bv, bv, bv};

    for (int kt = 0; kt < KT; ++kt) {
        int kbase = kt * 32;
        // A frag (16-bit A 16x32 layout): lane holds K = {8*khalf+0..7, 16+8*khalf+0..7}
        const bf16_t* pah = Ah + rowA * KPAD + kbase + khalf * 8;
        const bf16_t* pal = Al + rowA * KPAD + kbase + khalf * 8;
        v8bf ah0 = *(const v8bf*)(pah);
        v8bf ah1 = *(const v8bf*)(pah + 16);
        v8bf al0 = *(const v8bf*)(pal);
        v8bf al1 = *(const v8bf*)(pal + 16);
        v16bf ah = __builtin_shufflevector(ah0, ah1, 0,1,2,3,4,5,6,7,8,9,10,11,12,13,14,15);
        v16bf al = __builtin_shufflevector(al0, al1, 0,1,2,3,4,5,6,7,8,9,10,11,12,13,14,15);
        // B frag (bf16 B 32x16): lane holds N = lane%16, K = {8*khalf+0..7, 16+8*khalf+0..7}
        const bf16_t* pbh = Bh + (kbase + khalf * 8) * D_DIM + nt * 16 + m16;
        const bf16_t* pbl = Bl + (kbase + khalf * 8) * D_DIM + nt * 16 + m16;
        v16bf bh, bl;
        #pragma unroll
        for (int s = 0; s < 16; ++s) {
            int kk = (s < 8) ? s : s + 8;
            bh[s] = pbh[kk * D_DIM];
            bl[s] = pbl[kk * D_DIM];
        }

        acc = __builtin_amdgcn_wmma_f32_16x16x32_bf16(false, ah, false, bh, (short)0, acc, false, false);
        acc = __builtin_amdgcn_wmma_f32_16x16x32_bf16(false, ah, false, bl, (short)0, acc, false, false);
        acc = __builtin_amdgcn_wmma_f32_16x16x32_bf16(false, al, false, bh, (short)0, acc, false, false);
    }

    // C/D layout: VGPR j -> M = mt*16 + j + 8*khalf, N = nt*16 + lane%16
    int n = nt * 16 + m16;
    int mbase = mt * 16 + khalf * 8;
    #pragma unroll
    for (int j = 0; j < 8; ++j)
        C[(mbase + j) * D_DIM + n] = acc[j];
}

// ---- pairwise sigmoid-gated reduction + tanh, q-tiled (QT queries/block) ---
// grid.x = B * S/QT  (b and q-group), grid.y = 2 (d chunks of 256)
// Each x[k,d] / g1[k,d] load feeds QT sigmoid+FMA ops -> trans-pipe bound.
__global__ __launch_bounds__(256)
void context_reduce(const float* __restrict__ X, const int* __restrict__ masks,
                    const float* __restrict__ G0, const float* __restrict__ G1,
                    const float* __restrict__ lens, float* __restrict__ out) {
    __shared__ float smask[S_DIM];
    const float LOG2E = 1.4426950408889634f;
    int qgroups = S_DIM / QT;                 // 32
    int b  = blockIdx.x / qgroups;
    int q0 = (blockIdx.x % qgroups) * QT;
    int tid = threadIdx.x;
    if (tid < S_DIM) smask[tid] = (float)masks[b * S_DIM + tid];
    __syncthreads();

    int d = blockIdx.y * 256 + tid;
    if (d >= D_DIM) return;                   // after the only barrier: safe

    int rowbase = b * S_DIM;                  // flattened (b,s) row index base
    float g0q[QT];
    float acc[QT];
    #pragma unroll
    for (int j = 0; j < QT; ++j) {
        g0q[j] = G0[(rowbase + q0 + j) * D_DIM + d];
        acc[j] = 0.0f;
    }

    const float* xp = X  + rowbase * D_DIM + d;
    const float* gp = G1 + rowbase * D_DIM + d;

    #pragma unroll 2
    for (int k = 0; k < S_DIM; ++k) {
        float g1k = gp[k * D_DIM];
        float xmk = xp[k * D_DIM] * smask[k];
        #pragma unroll
        for (int j = 0; j < QT; ++j) {
            float t  = g0q[j] + g1k;
            float sg = __builtin_amdgcn_rcpf(1.0f + __builtin_amdgcn_exp2f(-t * LOG2E));
            acc[j] += xmk * sg;
        }
    }

    float rlen = __builtin_amdgcn_rcpf(lens[b]);
    #pragma unroll
    for (int j = 0; j < QT; ++j) {
        float ctx = acc[j] * smask[q0 + j] * rlen;
        // tanh(y) = 2*sigmoid(2y) - 1
        float th = 2.0f * __builtin_amdgcn_rcpf(1.0f + __builtin_amdgcn_exp2f(-2.0f * ctx * LOG2E)) - 1.0f;
        out[(rowbase + q0 + j) * D_DIM + d] = th;
    }
}

// ---------------------------------------------------------------------------
extern "C" void kernel_launch(void* const* d_in, const int* in_sizes, int n_in,
                              void* d_out, int out_size, void* d_ws, size_t ws_size,
                              hipStream_t stream) {
    const float* X     = (const float*)d_in[0];
    const int*   masks = (const int*)  d_in[1];
    const float* W0    = (const float*)d_in[2];
    const float* b0    = (const float*)d_in[3];
    const float* W1    = (const float*)d_in[4];
    const float* b1    = (const float*)d_in[5];
    float* out = (float*)d_out;

    // workspace carve-out (256B aligned regions), ~6.1 MB total
    char* ws = (char*)d_ws;
    size_t off = 0;
    auto carve = [&](size_t bytes) -> void* {
        off = (off + 255) & ~(size_t)255;
        void* p = ws + off;
        off += bytes;
        return p;
    };
    bf16_t* Xh  = (bf16_t*)carve((size_t)BQ * KPAD * 2);
    bf16_t* Xl  = (bf16_t*)carve((size_t)BQ * KPAD * 2);
    bf16_t* W0h = (bf16_t*)carve((size_t)KPAD * D_DIM * 2);
    bf16_t* W0l = (bf16_t*)carve((size_t)KPAD * D_DIM * 2);
    bf16_t* W1h = (bf16_t*)carve((size_t)KPAD * D_DIM * 2);
    bf16_t* W1l = (bf16_t*)carve((size_t)KPAD * D_DIM * 2);
    float*  G0  = (float*) carve((size_t)BQ * D_DIM * 4);
    float*  G1  = (float*) carve((size_t)BQ * D_DIM * 4);
    float*  len = (float*) carve(16);

    cvt_split_x <<<(BQ * KPAD + 255) / 256, 256, 0, stream>>>(X, Xh, Xl);
    cvt_split_wT<<<(KPAD * D_DIM + 255) / 256, 256, 0, stream>>>(W0, W0h, W0l);
    cvt_split_wT<<<(KPAD * D_DIM + 255) / 256, 256, 0, stream>>>(W1, W1h, W1l);
    seq_lengths <<<B_DIM, 256, 0, stream>>>(masks, len);

    gemm_bf16x3<<<(MT * NT) / 8, 256, 0, stream>>>(Xh, Xl, W0h, W0l, b0, G0);
    gemm_bf16x3<<<(MT * NT) / 8, 256, 0, stream>>>(Xh, Xl, W1h, W1l, b1, G1);

    context_reduce<<<dim3(B_DIM * (S_DIM / QT), 2), 256, 0, stream>>>(
        X, masks, G0, G1, len, out);
}
